// DifferentiableKMeansLayer_33655363732009
// MI455X (gfx1250) — matrix operand order repaired and checked by reading, not compiled, not run on hardware
//
#include <hip/hip_runtime.h>

typedef __attribute__((ext_vector_type(16))) _Float16 v16h;
typedef __attribute__((ext_vector_type(8)))  _Float16 v8h;
typedef __attribute__((ext_vector_type(4)))  _Float16 v4h;
typedef __attribute__((ext_vector_type(8)))  float    v8f;

#define D            64
#define K_CL         512
#define INV_T        10.0f
#define WG_THREADS   256
#define ROWS_PER_WG  128

// LDS layout (in elements)
#define CENT16_HALFS (K_CL * D)               // 32768 halfs = 64 KB: row-major f16 centroids (GEMM1 B)
#define B2FMT_HALFS  (8 * 2 * 4 * 32 * 16)    // 32768 halfs = 64 KB: pre-swizzled GEMM2 B fragments
#define CSQ_FLOATS   (K_CL)                   // 512 floats = 2 KB: |c|^2 * (1/T)
#define PSCR_HALFS   (8 * 16 * 64)            // 8192 halfs = 16 KB: per-wave P-tile transpose scratch
#define SMEM_BYTES   (CENT16_HALFS*2 + B2FMT_HALFS*2 + CSQ_FLOATS*4 + PSCR_HALFS*2)

union V16 { v16h v; v8h h8[2]; _Float16 e[16]; };

__global__ __launch_bounds__(WG_THREADS)
void kmeans_mixture_wmma(const float* __restrict__ x,
                         const float* __restrict__ cent,
                         float* __restrict__ out,
                         int nrows)
{
    extern __shared__ char smem_raw[];
    _Float16* cent16 = (_Float16*)smem_raw;
    _Float16* b2fmt  = cent16 + CENT16_HALFS;
    float*    csq10  = (float*)(b2fmt + B2FMT_HALFS);
    _Float16* pscr   = (_Float16*)(csq10 + CSQ_FLOATS);

    const int tid  = threadIdx.x;
    const int wave = tid >> 5;
    const int lane = tid & 31;
    const int l16  = lane & 15;   // column within 16-lane group
    const int lh   = lane >> 4;   // lane half (0: rows/K low, 1: rows/K high)

    const int rowbase = blockIdx.x * ROWS_PER_WG + wave * 16;
    const float* xr = x + (size_t)(rowbase + l16) * D;

    // Prefetch this wave's x rows; overlaps the LDS staging below.
    __builtin_prefetch(xr, 0, 0);
    __builtin_prefetch(xr + 32, 0, 0);

    // ---------------- LDS staging (once per workgroup) ----------------
    // f16 centroids, row-major [K][D]: GEMM1 B-fragments (B = C^T) are
    // two 8-half runs along D per lane (d = 8*lh.. and 16 + 8*lh..) -> 2x ds_load_b128 each.
    for (int i = tid; i < CENT16_HALFS; i += WG_THREADS)
        cent16[i] = (_Float16)cent[i];

    // Pre-swizzled GEMM2 B fragments with K-permutation k' = l16*4 + tile
    // (matches the packed ds_store_b64 layout of the P tile; softmax/mixture
    // are invariant to a consistent K permutation).
    for (int i = tid; i < B2FMT_HALFS; i += WG_THREADS) {
        int h  = i & 15;
        int ln = (i >> 4) & 31;
        int nt = (i >> 9) & 3;
        int f  = (i >> 11) & 1;
        int kc = i >> 12;
        int kkp  = (f << 5) + ((h >> 3) << 4) + ((ln >> 4) << 3) + (h & 7);        // permuted K within 64-chunk
        int orig = ((kkp & 3) << 4) + (kkp >> 2);          // original column in chunk
        int cl   = (kc << 6) + orig;                       // cluster index
        int n    = (nt << 4) + (ln & 15);                  // output feature index
        b2fmt[i] = (_Float16)cent[cl * D + n];
    }

    // |c|^2 / T in f32 (exact logit bias; centroids are L2-resident).
    for (int k = tid; k < K_CL; k += WG_THREADS) {
        float sum = 0.f;
        #pragma unroll
        for (int d = 0; d < D; ++d) { float v = cent[k * D + d]; sum = fmaf(v, v, sum); }
        csq10[k] = sum * INV_T;
    }
    __syncthreads();

    // ---------------- A fragments: 16 rows of x, f16, two 16x32 ----------------
    V16 A0, A1;
    #pragma unroll
    for (int j = 0; j < 8; ++j) {
        A0.e[j]     = (_Float16)xr[lh * 8 + j];
        A0.e[8 + j] = (_Float16)xr[16 + lh * 8 + j];
        A1.e[j]     = (_Float16)xr[32 + lh * 8 + j];
        A1.e[8 + j] = (_Float16)xr[48 + lh * 8 + j];
    }

    // All-ones B fragment: row-sum of P via WMMA (every output column = sum).
    V16 ONES;
    #pragma unroll
    for (int j = 0; j < 16; ++j) ONES.e[j] = (_Float16)1.0f;

    v8f acc[4] = {};     // 16 rows x 64 output features (f32)
    v8f accs  = {};      // 16 rows x (replicated) softmax denominator
    float mrow[8];
    #pragma unroll
    for (int r = 0; r < 8; ++r) mrow[r] = -1.0e30f;

    _Float16* ps = pscr + wave * (16 * 64);   // wave-private transpose scratch

    // ---------------- main loop: 8 chunks of 64 clusters ----------------
    for (int kc = 0; kc < 8; ++kc) {
        const int n0 = kc << 6;

        // cross = x . c for 4 tiles of 16 clusters (f32 accumulate, D=64)
        v8f cr[4];
        #pragma unroll
        for (int t = 0; t < 4; ++t) {
            const _Float16* cb = cent16 + (size_t)(n0 + t * 16 + l16) * D + lh * 8;
            V16 Bc0, Bc1;
            Bc0.h8[0] = *(const v8h*)(cb);       Bc0.h8[1] = *(const v8h*)(cb + 16);
            Bc1.h8[0] = *(const v8h*)(cb + 32);  Bc1.h8[1] = *(const v8h*)(cb + 48);
            v8f c = {};
            c = __builtin_amdgcn_wmma_f32_16x16x32_f16(false, A0.v, false, Bc0.v, (short)0, c, false, false);
            c = __builtin_amdgcn_wmma_f32_16x16x32_f16(false, A1.v, false, Bc1.v, (short)0, c, false, false);
            cr[t] = c;
        }

        // logits = (2*cross)/T - |c|^2/T   (|x|^2 cancels in the softmax)
        float l[4][8], vmax[8];
        #pragma unroll
        for (int t = 0; t < 4; ++t) {
            const float q = csq10[n0 + t * 16 + l16];
            #pragma unroll
            for (int r = 0; r < 8; ++r)
                l[t][r] = fmaf(2.0f * INV_T, cr[t][r], -q);
        }
        #pragma unroll
        for (int r = 0; r < 8; ++r)
            vmax[r] = fmaxf(fmaxf(l[0][r], l[1][r]), fmaxf(l[2][r], l[3][r]));

        // one row-max reduction per 64 clusters (16-lane groups = rows 0-7 | 8-15)
        #pragma unroll
        for (int off = 1; off < 16; off <<= 1) {
            #pragma unroll
            for (int r = 0; r < 8; ++r)
                vmax[r] = fmaxf(vmax[r], __shfl_xor(vmax[r], off, 32));
        }

        float scale[8];
        #pragma unroll
        for (int r = 0; r < 8; ++r) {
            float nm = fmaxf(mrow[r], vmax[r]);
            scale[r] = __expf(mrow[r] - nm);
            mrow[r]  = nm;
        }

        // p = exp(l - m), packed store: 4 tiles per row -> one ds_store_b64
        #pragma unroll
        for (int r = 0; r < 8; ++r) {
            v4h pk;
            #pragma unroll
            for (int t = 0; t < 4; ++t)
                pk[t] = (_Float16)__expf(l[t][r] - mrow[r]);
            *(v4h*)(ps + ((lh * 8 + r) << 6) + (l16 << 2)) = pk;   // k' = l16*4 + t
        }

        // online rescale of accumulators (row mapping matches C/D layout)
        #pragma unroll
        for (int r = 0; r < 8; ++r) {
            acc[0][r] *= scale[r];
            acc[1][r] *= scale[r];
            acc[2][r] *= scale[r];
            acc[3][r] *= scale[r];
            accs[r]   *= scale[r];
        }

        asm volatile("s_wait_dscnt 0" ::: "memory");

        // reload P tile as two 16x32 A fragments (permuted K)
        V16 P0, P1;
        P0.h8[0] = *(const v8h*)(ps + (l16 << 6) + lh * 8);
        P0.h8[1] = *(const v8h*)(ps + (l16 << 6) + 16 + lh * 8);
        P1.h8[0] = *(const v8h*)(ps + (l16 << 6) + 32 + lh * 8);
        P1.h8[1] = *(const v8h*)(ps + (l16 << 6) + 48 + lh * 8);

        // acc += P (16x64) x C (64x64); denominator via ones-matrix WMMA
        #pragma unroll
        for (int nt = 0; nt < 4; ++nt) {
            V16 B20, B21;
            const v8h* bp0 = (const v8h*)(b2fmt + ((((kc * 2 + 0) * 4 + nt) * 32 + lane) << 4));
            const v8h* bp1 = (const v8h*)(b2fmt + ((((kc * 2 + 1) * 4 + nt) * 32 + lane) << 4));
            B20.h8[0] = bp0[0];  B20.h8[1] = bp0[1];
            B21.h8[0] = bp1[0];  B21.h8[1] = bp1[1];
            acc[nt] = __builtin_amdgcn_wmma_f32_16x16x32_f16(false, P0.v, false, B20.v, (short)0, acc[nt], false, false);
            acc[nt] = __builtin_amdgcn_wmma_f32_16x16x32_f16(false, P1.v, false, B21.v, (short)0, acc[nt], false, false);
        }
        accs = __builtin_amdgcn_wmma_f32_16x16x32_f16(false, P0.v, false, ONES.v, (short)0, accs, false, false);
        accs = __builtin_amdgcn_wmma_f32_16x16x32_f16(false, P1.v, false, ONES.v, (short)0, accs, false, false);
    }

    // ---------------- epilogue: out = acc / sum ----------------
    float inv[8];
    #pragma unroll
    for (int r = 0; r < 8; ++r) inv[r] = 1.0f / accs[r];

    float* orow = out + (size_t)(rowbase + lh * 8) * D + l16;
    #pragma unroll
    for (int r = 0; r < 8; ++r) {
        #pragma unroll
        for (int nt = 0; nt < 4; ++nt)
            orow[(size_t)r * D + nt * 16] = acc[nt][r] * inv[r];
    }
    (void)nrows;
}

extern "C" void kernel_launch(void* const* d_in, const int* in_sizes, int n_in,
                              void* d_out, int out_size, void* d_ws, size_t ws_size,
                              hipStream_t stream) {
    const float* x    = (const float*)d_in[0];   // [16, 8192, 64] f32
    const float* cent = (const float*)d_in[1];   // [512, 64] f32
    float* out = (float*)d_out;                  // [16, 8192, 64] f32

    int nrows   = in_sizes[0] / D;               // 131072
    int nblocks = nrows / ROWS_PER_WG;           // 1024

    kmeans_mixture_wmma<<<nblocks, WG_THREADS, SMEM_BYTES, stream>>>(x, cent, out, nrows);
}
